// HebbyRNN_81827716924021
// MI455X (gfx1250) — compile-verified, matrix operand order checked by reading, not run on hardware
//
#include <hip/hip_runtime.h>
#include <hip/hip_bf16.h>
#include <cmath>

typedef __attribute__((ext_vector_type(16))) __bf16 v16bf;
typedef __attribute__((ext_vector_type(8)))  __bf16 v8bf;
typedef __attribute__((ext_vector_type(8)))  float  v8f;
typedef __attribute__((ext_vector_type(4)))  int    v4i;

#define BATCH 4096
#define KDIM  2560   // inner dim for every GEMM
#define BM 256       // block tile M
#define BN 128       // block tile N
#define BK 32        // K step
#define LDP 40       // LDS row pitch in bf16 elements (32 data + 8 pad)

// --- CDNA5 async global->LDS copy (direct-to-LDS, ASYNCcnt tracked) --------
#if defined(__has_builtin)
#if __has_builtin(__builtin_amdgcn_global_load_async_to_lds_b128)
#define HAVE_ASYNC 1
#endif
#endif
#ifndef HAVE_ASYNC
#define HAVE_ASYNC 0
#endif

#if HAVE_ASYNC
typedef __attribute__((address_space(1))) v4i g_v4i;   // global
typedef __attribute__((address_space(3))) v4i l_v4i;   // LDS

__device__ __forceinline__ void async_copy16(const __bf16* g, __bf16* l) {
    __builtin_amdgcn_global_load_async_to_lds_b128(
        (g_v4i*)g, (l_v4i*)l, 0, 0);
}
#endif

__device__ __forceinline__ void wait_async0() {
#if __has_builtin(__builtin_amdgcn_s_wait_asynccnt)
    __builtin_amdgcn_s_wait_asynccnt(0);
#else
    asm volatile("s_wait_asynccnt 0x0" ::: "memory");
#endif
}

__device__ __forceinline__ __bf16 to_bf16(float f) {
    unsigned int u = __builtin_bit_cast(unsigned int, f);
    u += 0x7FFFu + ((u >> 16) & 1u);            // round-to-nearest-even
    return __builtin_bit_cast(__bf16, (unsigned short)(u >> 16));
}

// ---------------------------------------------------------------------------
// Pack concat(x [B,512], h [B,2048]) -> bf16 combined [B,2560]
// ---------------------------------------------------------------------------
__global__ __launch_bounds__(256) void pack_kernel(
    const float* __restrict__ x, const float* __restrict__ h,
    __bf16* __restrict__ out)
{
    int i = blockIdx.x * blockDim.x + threadIdx.x;          // chunk id (8 elts)
    int row = i / 320;
    int col = (i % 320) * 8;
    const float* src = (col < 512) ? (x + (size_t)row * 512 + col)
                                   : (h + (size_t)row * 2048 + (col - 512));
    const float4* p = reinterpret_cast<const float4*>(src);
    float4 a = p[0], b = p[1];
    v8bf o;
    o[0]=to_bf16(a.x); o[1]=to_bf16(a.y); o[2]=to_bf16(a.z); o[3]=to_bf16(a.w);
    o[4]=to_bf16(b.x); o[5]=to_bf16(b.y); o[6]=to_bf16(b.z); o[7]=to_bf16(b.w);
    reinterpret_cast<v8bf*>(out)[i] = o;
}

// ---------------------------------------------------------------------------
// f32 -> bf16 weight conversion, 8 elements per thread
// ---------------------------------------------------------------------------
__global__ __launch_bounds__(256) void cvt_kernel(
    const float* __restrict__ in, __bf16* __restrict__ out, int n8)
{
    int i = blockIdx.x * blockDim.x + threadIdx.x;
    if (i >= n8) return;
    const float4* p = reinterpret_cast<const float4*>(in) + 2 * (size_t)i;
    float4 a = p[0], b = p[1];
    v8bf o;
    o[0]=to_bf16(a.x); o[1]=to_bf16(a.y); o[2]=to_bf16(a.z); o[3]=to_bf16(a.w);
    o[4]=to_bf16(b.x); o[5]=to_bf16(b.y); o[6]=to_bf16(b.z); o[7]=to_bf16(b.w);
    reinterpret_cast<v8bf*>(out)[i] = o;
}

// ---------------------------------------------------------------------------
// WMMA GEMM:  Out[m,n] = act( sum_k A[m,k]*W[n,k] + bias[n] )
//   A : bf16 [4096, 2560] row-major;  W : bf16 [N, 2560] row-major
//   MODE 0: ReLU -> bf16, MODE 1: tanh -> f32, MODE 2: none -> f32
// Block: 256 threads (8 waves, 4x2), tile 256x128, wave tile 64x64,
// double-buffered LDS, async global->LDS copies when available.
// ---------------------------------------------------------------------------
template<int MODE>
__global__ __launch_bounds__(256) void gemm_bf16_kernel(
    const __bf16* __restrict__ A,
    const __bf16* __restrict__ W,
    const float*  __restrict__ bias,
    void* __restrict__ Out, int ldOut)
{
    __shared__ __bf16 sA[2][BM * LDP];   // 2 * 256*40*2B = 40960 B
    __shared__ __bf16 sB[2][BN * LDP];   // 2 * 128*40*2B = 20480 B

    const int tid  = threadIdx.x;
    const int lane = tid & 31;
    const int wave = tid >> 5;
    const int wm   = wave >> 1;      // 0..3  (64 rows each)
    const int wn   = wave & 1;       // 0..1  (64 cols each)
    const int mBase = blockIdx.y * BM;
    const int nBase = blockIdx.x * BN;

    // global->LDS mapping: chunk = 8 bf16 (16 B); A: 4 chunks/thr, B: 2
    const int ar = tid >> 2;                 // rows 0..63
    const int ac = (tid & 3) * 8;

    const __bf16* gA[4];
    const __bf16* gB[2];
#pragma unroll
    for (int q = 0; q < 4; ++q)
        gA[q] = A + (size_t)(mBase + ar + 64 * q) * KDIM + ac;
#pragma unroll
    for (int q = 0; q < 2; ++q)
        gB[q] = W + (size_t)(nBase + ar + 64 * q) * KDIM + ac;

    v8f c[4][4];
#pragma unroll
    for (int i = 0; i < 4; ++i)
#pragma unroll
        for (int j = 0; j < 4; ++j)
            c[i][j] = (v8f)(0.0f);

#if HAVE_ASYNC
    // prologue: async copy tile 0 directly into LDS buffer 0
#pragma unroll
    for (int q = 0; q < 4; ++q)
        async_copy16(gA[q], &sA[0][(ar + 64 * q) * LDP + ac]);
#pragma unroll
    for (int q = 0; q < 2; ++q)
        async_copy16(gB[q], &sB[0][(ar + 64 * q) * LDP + ac]);
#else
    v8bf ra[4], rb[2];
#pragma unroll
    for (int q = 0; q < 4; ++q) ra[q] = *(const v8bf*)gA[q];
#pragma unroll
    for (int q = 0; q < 2; ++q) rb[q] = *(const v8bf*)gB[q];
#pragma unroll
    for (int q = 0; q < 4; ++q) *(v8bf*)&sA[0][(ar + 64 * q) * LDP + ac] = ra[q];
#pragma unroll
    for (int q = 0; q < 2; ++q) *(v8bf*)&sB[0][(ar + 64 * q) * LDP + ac] = rb[q];
#endif

    const int halfOff = (lane >> 4) * 8;   // K sub-block select per ISA layout
    const int rowSel  = lane & 15;

    int buf = 0;
    const int NK = KDIM / BK;   // 80
    for (int kt = 0; kt < NK; ++kt) {
#if HAVE_ASYNC
        wait_async0();                    // tile kt landed in LDS[buf]
#endif
        __syncthreads();

        if (kt + 1 < NK) {
            const int ko = (kt + 1) * BK;
            const int nb = buf ^ 1;
#if HAVE_ASYNC
#pragma unroll
            for (int q = 0; q < 4; ++q)
                async_copy16(gA[q] + ko, &sA[nb][(ar + 64 * q) * LDP + ac]);
#pragma unroll
            for (int q = 0; q < 2; ++q)
                async_copy16(gB[q] + ko, &sB[nb][(ar + 64 * q) * LDP + ac]);
#else
#pragma unroll
            for (int q = 0; q < 4; ++q) ra[q] = *(const v8bf*)(gA[q] + ko);
#pragma unroll
            for (int q = 0; q < 2; ++q) rb[q] = *(const v8bf*)(gB[q] + ko);
            (void)nb;
#endif
        }

        // fragment loads (two ds_load_b128 each, per documented 16-bit layout)
        v16bf af[4], bfr[4];
#pragma unroll
        for (int i = 0; i < 4; ++i) {
            const __bf16* p = &sA[buf][(wm * 64 + i * 16 + rowSel) * LDP + halfOff];
            union { v16bf v; v8bf h[2]; } t;
            t.h[0] = *(const v8bf*)p;
            t.h[1] = *(const v8bf*)(p + 16);
            af[i] = t.v;
        }
#pragma unroll
        for (int j = 0; j < 4; ++j) {
            const __bf16* p = &sB[buf][(wn * 64 + j * 16 + rowSel) * LDP + halfOff];
            union { v16bf v; v8bf h[2]; } t;
            t.h[0] = *(const v8bf*)p;
            t.h[1] = *(const v8bf*)(p + 16);
            bfr[j] = t.v;
        }

#pragma unroll
        for (int i = 0; i < 4; ++i)
#pragma unroll
            for (int j = 0; j < 4; ++j)
                c[i][j] = __builtin_amdgcn_wmma_f32_16x16x32_bf16(
                    false, af[i], false, bfr[j], (short)0, c[i][j], false, false);

#if !HAVE_ASYNC
        if (kt + 1 < NK) {
            const int nb = buf ^ 1;
#pragma unroll
            for (int q = 0; q < 4; ++q) *(v8bf*)&sA[nb][(ar + 64 * q) * LDP + ac] = ra[q];
#pragma unroll
            for (int q = 0; q < 2; ++q) *(v8bf*)&sB[nb][(ar + 64 * q) * LDP + ac] = rb[q];
        }
#endif
        buf ^= 1;
    }

    // epilogue: C/D layout — VGPR r, lanes 0-15: M=r, lanes 16-31: M=8+r, N=lane%16
    const int rowHalf = (lane >> 4) * 8;
    const int colIn   = lane & 15;
#pragma unroll
    for (int j = 0; j < 4; ++j) {
        const int col = nBase + wn * 64 + j * 16 + colIn;
        const float bv = bias[col];
#pragma unroll
        for (int i = 0; i < 4; ++i) {
            const int rbase = mBase + wm * 64 + i * 16 + rowHalf;
#pragma unroll
            for (int r = 0; r < 8; ++r) {
                float v = c[i][j][r] + bv;
                const size_t off = (size_t)(rbase + r) * ldOut + col;
                if (MODE == 0) {
                    v = v > 0.0f ? v : 0.0f;
                    ((__bf16*)Out)[off] = to_bf16(v);
                } else if (MODE == 1) {
                    ((float*)Out)[off] = tanhf(v);
                } else {
                    ((float*)Out)[off] = v;
                }
            }
        }
    }
}

// ---------------------------------------------------------------------------
extern "C" void kernel_launch(void* const* d_in, const int* in_sizes, int n_in,
                              void* d_out, int out_size, void* d_ws, size_t ws_size,
                              hipStream_t stream) {
    const float* x        = (const float*)d_in[0];
    const float* h        = (const float*)d_in[1];
    const float* W_layers = (const float*)d_in[2];
    const float* b_layers = (const float*)d_in[3];
    const float* W_i2h    = (const float*)d_in[4];
    const float* b_i2h    = (const float*)d_in[5];
    const float* W_i2o    = (const float*)d_in[6];
    const float* b_i2o    = (const float*)d_in[7];
    float* out = (float*)d_out;   // [4096*512] output, then [4096*2048] hidden

    char* ws = (char*)d_ws;
    __bf16* actA = (__bf16*)ws;  ws += (size_t)BATCH * KDIM * 2;
    __bf16* actB = (__bf16*)ws;  ws += (size_t)BATCH * KDIM * 2;
    __bf16* wL   = (__bf16*)ws;  ws += (size_t)4 * KDIM * KDIM * 2;
    __bf16* wIH  = (__bf16*)ws;  ws += (size_t)2048 * KDIM * 2;
    __bf16* wIO  = (__bf16*)ws;

    // pack combined = concat(x,h) as bf16
    {
        int nChunks = BATCH * (KDIM / 8);
        pack_kernel<<<nChunks / 256, 256, 0, stream>>>(x, h, actA);
    }
    // convert weights to bf16
    {
        int n8 = (int)((size_t)4 * KDIM * KDIM / 8);
        cvt_kernel<<<(n8 + 255) / 256, 256, 0, stream>>>(W_layers, wL, n8);
        n8 = (int)((size_t)2048 * KDIM / 8);
        cvt_kernel<<<(n8 + 255) / 256, 256, 0, stream>>>(W_i2h, wIH, n8);
        n8 = (int)((size_t)512 * KDIM / 8);
        cvt_kernel<<<(n8 + 255) / 256, 256, 0, stream>>>(W_i2o, wIO, n8);
    }

    dim3 blk(256);
    // 4 square Hebbian layers, ReLU, ping-pong bf16 activations
    for (int l = 0; l < 4; ++l) {
        const __bf16* src = (l & 1) ? actB : actA;
        __bf16*       dst = (l & 1) ? actA : actB;
        gemm_bf16_kernel<0><<<dim3(KDIM / BN, BATCH / BM), blk, 0, stream>>>(
            src, wL + (size_t)l * KDIM * KDIM, b_layers + (size_t)l * KDIM,
            dst, KDIM);
    }
    // after 4 layers (A->B->A->B->A), final activations are in actA
    gemm_bf16_kernel<1><<<dim3(2048 / BN, BATCH / BM), blk, 0, stream>>>(
        actA, wIH, b_i2h, out + (size_t)BATCH * 512, 2048);
    gemm_bf16_kernel<2><<<dim3(512 / BN, BATCH / BM), blk, 0, stream>>>(
        actA, wIO, b_i2o, out, 512);
}
